// ConvMemory_4080218931756
// MI455X (gfx1250) — matrix operand order repaired and checked by reading, not compiled, not run on hardware
//
#include <hip/hip_runtime.h>

// ---------------------------------------------------------------------------
// ConvMemory: out[b,o,y,x] = sum_k memory[k,o] * softmax_k(unfold3x3(x)[b,k,y,x])
// x: [8,64,128,128] f32, memory: [576,128] f32, out: [8,128,128,128] f32
//
// Strategy (see analysis): fused softmax -> f16 B-fragments in LDS, memory
// pre-packed to f16 A-fragments in d_ws, projection via v_wmma_f32_16x16x32_f16
// with f32 accumulation. Memory-bound (~100MB HBM); WMMA keeps GEMM off VALU.
// ---------------------------------------------------------------------------

typedef _Float16    v16h __attribute__((ext_vector_type(16)));
typedef float       v8f  __attribute__((ext_vector_type(8)));
typedef unsigned int v8u __attribute__((ext_vector_type(8)));

#define C_IN   64
#define HH     128
#define WW     128
#define NB     8
#define OC     128
#define KTOT   576           // 64 * 3 * 3
#define KSTEPS 18            // 576 / 32
#define NT     8             // 128 positions / 16
#define MT     8             // 128 out-ch / 16

// ---------------------------------------------------------------------------
// Kernel 1: pack memory[576,128] f32 into f16 WMMA A-fragments in d_ws.
// Layout: ws[((mt*18 + ks)*32 + lane)*8 + v]  (dwords, 2 f16 each) = 144 KB.
// A-fragment (16-bit A 16x32, ISA 7.12.2): lane L, vgpr v holds
//   M = L&15,  K = (v>=4?16:0) + (L>=16?8:0) + 2*(v&3)  (and K+1 in hi half)
// A[m][k] = memory[32*ks + k][16*mt + m]   (A = memory^T tile)
// ---------------------------------------------------------------------------
__global__ __launch_bounds__(256) void convmem_pack_A(
    const float* __restrict__ mem, unsigned int* __restrict__ ws) {
  int tid = blockIdx.x * 256 + threadIdx.x;      // 144 blocks * 256 = 36864
  int v    = tid & 7;
  int lane = (tid >> 3) & 31;
  int rest = tid >> 8;
  int ks   = rest % KSTEPS;
  int mt   = rest / KSTEPS;
  int m    = lane & 15;
  int klo  = ((v >= 4) ? 16 : 0) + ((lane >= 16) ? 8 : 0) + 2 * (v & 3);
  int k    = ks * 32 + klo;
  int o    = mt * 16 + m;
  _Float16 h0 = (_Float16)mem[k * OC + o];
  _Float16 h1 = (_Float16)mem[(k + 1) * OC + o];
  unsigned int u0 = __builtin_bit_cast(unsigned short, h0);
  unsigned int u1 = __builtin_bit_cast(unsigned short, h1);
  ws[tid] = (u1 << 16) | u0;
}

// ---------------------------------------------------------------------------
// Kernel 2: one workgroup per (batch, image row y): 128 positions.
// Phase 1: softmax probs as f16 B-fragments in LDS.
//   B-fragment (16-bit B 32x16): lane L, vgpr v holds
//     N = L&15,  K = (v>=4?16:0) + (L>=16?8:0) + 2*(v&3)  (and K+1 in hi half)
//   LDS layout pB[((nt*18 + ks)*32 + lane)*8 + v]  -> ds_load_b128 pairs.
// Phase 2: wave w = out-channel tile; 18 K-steps x 8 N-tiles of WMMA.
// ---------------------------------------------------------------------------
__global__ __launch_bounds__(256) void convmem_main(
    const float* __restrict__ x, const unsigned int* __restrict__ wsA,
    float* __restrict__ out) {
  __shared__ __attribute__((aligned(32))) unsigned int pB[NT * KSTEPS * 32 * 8];
  __shared__ float ssum[256];

  const int t    = threadIdx.x;
  const int b    = blockIdx.x >> 7;
  const int y    = blockIdx.x & 127;

  // ---------------- Phase 1: softmax -> f16 B fragments --------------------
  const int pos    = t & 127;        // spatial column
  const int hc     = t >> 7;         // channel half: c in [32*hc, 32*hc+32)
  const int ntile1 = pos >> 4;
  const int lane15 = pos & 15;
  const float* xb  = x + (size_t)b * C_IN * HH * WW;

  float sum = 0.f;
  {
    int c = hc * 32, kh = 0, kw = 0;
    const int kbase = hc * 288;
#define CM_ADV() do { if (++kw == 3) { kw = 0; if (++kh == 3) { kh = 0; ++c; } } } while (0)
    for (int kk = 0; kk < 288; kk += 2) {
      float e0, e1;
      {
        int yy = y + kh - 1, xx = pos + kw - 1;
        float val = ((unsigned)yy < 128u && (unsigned)xx < 128u)
                        ? xb[(c * HH + yy) * WW + xx] : 0.f;
        e0 = __expf(val);
      }
      CM_ADV();
      {
        int yy = y + kh - 1, xx = pos + kw - 1;
        float val = ((unsigned)yy < 128u && (unsigned)xx < 128u)
                        ? xb[(c * HH + yy) * WW + xx] : 0.f;
        e1 = __expf(val);
      }
      CM_ADV();
      sum += e0 + e1;
      int k      = kbase + kk;            // even
      int ks     = k >> 5;
      int klocal = k & 31;
      int half   = (klocal >> 3) & 1;     // lane half (K&8)
      int v2     = ((klocal >> 4) << 2) | ((klocal & 7) >> 1);    // fragment vgpr index
      _Float16 h0 = (_Float16)e0, h1 = (_Float16)e1;
      unsigned int u0 = __builtin_bit_cast(unsigned short, h0);
      unsigned int u1 = __builtin_bit_cast(unsigned short, h1);
      pB[(((ntile1 * KSTEPS) + ks) * 32 + lane15 + 16 * half) * 8 + v2] =
          (u1 << 16) | u0;
    }
#undef CM_ADV
  }

  ssum[t] = sum;
  __syncthreads();
  const float inv = 1.0f / (ssum[pos] + ssum[pos + 128]);

  // Rescale exactly the dwords this thread wrote (no cross-thread deps).
  for (int ksl = 0; ksl < 9; ++ksl) {
    int ks = hc * 9 + ksl;
    unsigned int* bp = &pB[((ntile1 * KSTEPS + ks) * 32 + lane15) * 8];
#pragma unroll
    for (int hseg = 0; hseg < 2; ++hseg) {
      unsigned int* q = bp + hseg * 16 * 8;
#pragma unroll
      for (int v2 = 0; v2 < 8; ++v2) {
        unsigned int u = q[v2];
        _Float16 lo = __builtin_bit_cast(_Float16, (unsigned short)(u & 0xffffu));
        _Float16 hi = __builtin_bit_cast(_Float16, (unsigned short)(u >> 16));
        lo = (_Float16)((float)lo * inv);
        hi = (_Float16)((float)hi * inv);
        unsigned int ulo = __builtin_bit_cast(unsigned short, lo);
        unsigned int uhi = __builtin_bit_cast(unsigned short, hi);
        q[v2] = (uhi << 16) | ulo;
      }
    }
  }
  __syncthreads();

  // ---------------- Phase 2: WMMA projection -------------------------------
  const int wv   = t >> 5;       // wave id == out-channel tile mt
  const int lane = t & 31;

  v8f acc[NT] = {};
  const unsigned int* Abase = wsA + ((size_t)(wv * KSTEPS) * 32 + lane) * 8;

  for (int ks = 0; ks < KSTEPS; ++ks) {
    v8u au = *(const v8u*)(Abase + (size_t)ks * 32 * 8);
    v16h a = __builtin_bit_cast(v16h, au);
#pragma unroll
    for (int nt = 0; nt < NT; ++nt) {
      v8u bu = *(const v8u*)&pB[((nt * KSTEPS + ks) * 32 + lane) * 8];
      v16h bf = __builtin_bit_cast(v16h, bu);
      acc[nt] = __builtin_amdgcn_wmma_f32_16x16x32_f16(
          false, a, false, bf, (short)0, acc[nt], false, false);
    }
  }

  // C/D layout: lane<16 -> M=r, N=lane; lane>=16 -> M=8+r, N=lane-16
  const int mrow = 16 * wv + ((lane >= 16) ? 8 : 0);
  const int ncol = lane & 15;
#pragma unroll
  for (int nt = 0; nt < NT; ++nt) {
    int xw = nt * 16 + ncol;
#pragma unroll
    for (int r = 0; r < 8; ++r) {
      int o = mrow + r;
      out[(((size_t)b * OC + o) * HH + y) * WW + xw] = acc[nt][r];
    }
  }
}

// ---------------------------------------------------------------------------
extern "C" void kernel_launch(void* const* d_in, const int* in_sizes, int n_in,
                              void* d_out, int out_size, void* d_ws, size_t ws_size,
                              hipStream_t stream) {
  (void)in_sizes; (void)n_in; (void)out_size; (void)ws_size;
  const float* x   = (const float*)d_in[0];   // [8,64,128,128]
  const float* mem = (const float*)d_in[1];   // [576,128]
  float* out       = (float*)d_out;           // [8,128,128,128]
  unsigned int* ws = (unsigned int*)d_ws;     // 144 KB of A fragments

  convmem_pack_A<<<144, 256, 0, stream>>>(mem, ws);               // 36864 thr
  convmem_main<<<NB * HH, 256, 0, stream>>>(x, ws, out);          // 1024 WGs
}
